// TransformerBlock_33560874451474
// MI455X (gfx1250) — compile-verified, matrix operand order checked by reading, not run on hardware
//
#include <hip/hip_runtime.h>

// ---------------------------------------------------------------------------
// Problem constants (from the reference)
// ---------------------------------------------------------------------------
#define BB   4
#define TT   1024
#define DD   1024
#define HH   4096
#define NHD  16
#define DHD  64
#define EE   8
#define NN   (BB * TT)          // 4096 tokens

typedef __attribute__((ext_vector_type(16))) __bf16 bf16x16;
typedef __attribute__((ext_vector_type(8)))  float  floatx8;

union FragU { bf16x16 v; uint4 q[2]; };

// ---------------------------------------------------------------------------
// bf16 helpers (round-to-nearest-even)
// ---------------------------------------------------------------------------
__device__ __forceinline__ __bf16 f2bf(float x) {
    union { float f; unsigned u; } a; a.f = x;
    unsigned r = (a.u + 0x7FFFu + ((a.u >> 16) & 1u)) >> 16;
    union { unsigned short s; __bf16 b; } o; o.s = (unsigned short)r;
    return o.b;
}
__device__ __forceinline__ float bf2f(__bf16 x) {
    union { unsigned short s; __bf16 b; } i; i.b = x;
    union { float f; unsigned u; } o; o.u = ((unsigned)i.s) << 16;
    return o.f;
}

// ---------------------------------------------------------------------------
// WMMA wrapper: D = A(16x32 bf16) * B(32x16 bf16) + C(16x16 f32)
// ---------------------------------------------------------------------------
__device__ __forceinline__ floatx8 wmma_bf16(bf16x16 a, bf16x16 b, floatx8 c) {
    return __builtin_amdgcn_wmma_f32_16x16x32_bf16(
        false, a, false, b, (short)0, c, false, false);
}

// Load a 16-lane-row fragment from a row-major LDS matrix [rows][ld] bf16.
// A-frag:  row0 = tile M offset (lane%16 selects M row).
// B-frag:  the matrix must be stored transposed (rows = N columns), contraction
//          index contiguous, so the same loader works (matches CDNA5 layout:
//          lanes<16 hold K {0..7,16..23}, lanes>=16 hold K {8..15,24..31}).
__device__ __forceinline__ bf16x16 load_frag(const __bf16* base, int row0,
                                             int ld, int k0) {
    int l    = threadIdx.x & 31;
    int r    = row0 + (l & 15);
    int hsel = (l < 16) ? 0 : 8;
    const __bf16* p = base + (size_t)r * ld + k0 + hsel;
    FragU f;
    f.q[0] = *(const uint4*)(p);
    f.q[1] = *(const uint4*)(p + 16);
    return f.v;
}

// ---------------------------------------------------------------------------
// LayerNorm (fp32 in -> bf16 out), one block per row of D=1024
// ---------------------------------------------------------------------------
__global__ __launch_bounds__(256) void layernorm_bf16(
    const float* __restrict__ x, const float* __restrict__ scale,
    const float* __restrict__ shift, __bf16* __restrict__ out, int D)
{
    int row = blockIdx.x;
    const float* xr = x + (size_t)row * D;
    float s = 0.f, s2 = 0.f;
    for (int d = threadIdx.x; d < D; d += 256) {
        float v = xr[d]; s += v; s2 += v * v;
    }
    for (int m = 16; m >= 1; m >>= 1) {
        s  += __shfl_xor(s,  m, 32);
        s2 += __shfl_xor(s2, m, 32);
    }
    __shared__ float red[2][8];
    if ((threadIdx.x & 31) == 0) {
        red[0][threadIdx.x >> 5] = s;
        red[1][threadIdx.x >> 5] = s2;
    }
    __syncthreads();
    s = 0.f; s2 = 0.f;
    for (int w = 0; w < 8; w++) { s += red[0][w]; s2 += red[1][w]; }
    float mean = s / D;
    float var  = s2 / D - mean * mean;
    float inv  = rsqrtf(var + 1e-5f);
    for (int d = threadIdx.x; d < D; d += 256) {
        float v = (xr[d] - mean) * inv * scale[d] + shift[d];
        out[(size_t)row * D + d] = f2bf(v);
    }
}

// ---------------------------------------------------------------------------
// fp32 [R][C] -> bf16 transposed [C][R], grid.z selects the expert slice
// ---------------------------------------------------------------------------
__global__ __launch_bounds__(256) void cvt_transpose(
    const float* __restrict__ in, __bf16* __restrict__ out, int R, int C)
{
    size_t mat = (size_t)R * C;
    const float* ip = in  + (size_t)blockIdx.z * mat;
    __bf16*      op = out + (size_t)blockIdx.z * mat;
    size_t idx = (size_t)blockIdx.x * 256 + threadIdx.x;
    if (idx >= mat) return;
    size_t c = idx / R, r = idx % R;         // output element [c][r]
    op[idx] = f2bf(ip[r * C + c]);
}

// ---------------------------------------------------------------------------
// Main WMMA GEMM: C[M x Nc] = A[M x Kc](bf16) * BT[Nc x Kc](bf16)^T
// Block tile 128x64, BK=32, 8 waves each doing a 32x32 (2x2 WMMA) sub-tile.
// Fusions: +bias[col], +resid[row][col], *rowscale[row] (with all-zero tile
// skip), accumulate-into outF, and/or bf16 store outB.
// ---------------------------------------------------------------------------
#define GBM 128
#define GBN 64
#define GBK 32

__global__ __launch_bounds__(256) void gemm_bf16(
    const __bf16* __restrict__ A, const __bf16* __restrict__ BT,
    const float* __restrict__ bias, const float* __restrict__ resid,
    const float* __restrict__ rowscale,
    float* __restrict__ outF, __bf16* __restrict__ outB,
    int M, int Nc, int Kc, int accum)
{
    __shared__ __align__(16) __bf16 As[GBM][GBK];
    __shared__ __align__(16) __bf16 Bs[GBN][GBK];
    __shared__ int skipFlag;

    int m0 = blockIdx.x * GBM, n0 = blockIdx.y * GBN;
    int tid = threadIdx.x;

    if (rowscale) {                  // MoE: skip tile if no token routed here
        if (tid == 0) skipFlag = 0;
        __syncthreads();
        if (tid < GBM && rowscale[m0 + tid] != 0.f) atomicOr(&skipFlag, 1);
        __syncthreads();
        if (!skipFlag) return;
    }

    int wave = tid >> 5, lane = tid & 31;
    int wm = wave >> 1, wn = wave & 1;

    floatx8 acc[2][2] = {};
    int rA = tid >> 1, hA = (tid & 1) * 16;      // A loader: 16 bf16 / thread
    int rB = tid >> 2, qB = (tid & 3) * 8;       // B loader: 8 bf16 / thread

    for (int kt = 0; kt < Kc; kt += GBK) {
        const uint4* sa = (const uint4*)(A + (size_t)(m0 + rA) * Kc + kt + hA);
        *(uint4*)&As[rA][hA]     = sa[0];
        *(uint4*)&As[rA][hA + 8] = sa[1];
        *(uint4*)&Bs[rB][qB] = *(const uint4*)(BT + (size_t)(n0 + rB) * Kc + kt + qB);
        if (kt + GBK < Kc)   // hint next A tile into cache (global_prefetch_b8)
            __builtin_prefetch(A + (size_t)(m0 + rA) * Kc + kt + GBK, 0, 1);
        __syncthreads();

        bf16x16 af0 = load_frag(&As[0][0], wm * 32,      GBK, 0);
        bf16x16 af1 = load_frag(&As[0][0], wm * 32 + 16, GBK, 0);
        bf16x16 bf0 = load_frag(&Bs[0][0], wn * 32,      GBK, 0);
        bf16x16 bf1 = load_frag(&Bs[0][0], wn * 32 + 16, GBK, 0);
        acc[0][0] = wmma_bf16(af0, bf0, acc[0][0]);
        acc[0][1] = wmma_bf16(af0, bf1, acc[0][1]);
        acc[1][0] = wmma_bf16(af1, bf0, acc[1][0]);
        acc[1][1] = wmma_bf16(af1, bf1, acc[1][1]);
        __syncthreads();
    }

    int hsel = (lane < 16) ? 0 : 8;
    int cl   = lane & 15;
    for (int i = 0; i < 2; i++)
        for (int j = 0; j < 2; j++) {
            int baseRow = m0 + wm * 32 + i * 16 + hsel;
            int baseCol = n0 + wn * 32 + j * 16 + cl;
            float bcol = bias ? bias[baseCol] : 0.f;
            for (int r = 0; r < 8; r++) {
                int row = baseRow + r;
                size_t off = (size_t)row * Nc + baseCol;
                float v = acc[i][j][r] + bcol;
                if (resid)    v += resid[off];
                if (rowscale) v *= rowscale[row];
                if (outF) { if (accum) outF[off] += v; else outF[off] = v; }
                if (outB) outB[off] = f2bf(v);
            }
        }
}

// ---------------------------------------------------------------------------
// Fused MoE up+gate: act = silu(A*W1T + b1) * (A*W2T + b2)  (bf16 out)
// ---------------------------------------------------------------------------
__global__ __launch_bounds__(256) void gemm_dual_silu(
    const __bf16* __restrict__ A,
    const __bf16* __restrict__ B1T, const __bf16* __restrict__ B2T,
    const float* __restrict__ b1, const float* __restrict__ b2,
    const float* __restrict__ rowscale,
    __bf16* __restrict__ outB, int M, int Nc, int Kc)
{
    __shared__ __align__(16) __bf16 As [GBM][GBK];
    __shared__ __align__(16) __bf16 B1s[GBN][GBK];
    __shared__ __align__(16) __bf16 B2s[GBN][GBK];
    __shared__ int skipFlag;

    int m0 = blockIdx.x * GBM, n0 = blockIdx.y * GBN;
    int tid = threadIdx.x;

    if (tid == 0) skipFlag = 0;
    __syncthreads();
    if (tid < GBM && rowscale[m0 + tid] != 0.f) atomicOr(&skipFlag, 1);
    __syncthreads();
    if (!skipFlag) return;    // rows scaled by 0 later -> safe to skip

    int wave = tid >> 5, lane = tid & 31;
    int wm = wave >> 1, wn = wave & 1;

    floatx8 acc1[2][2] = {}, acc2[2][2] = {};
    int rA = tid >> 1, hA = (tid & 1) * 16;
    int rB = tid >> 2, qB = (tid & 3) * 8;

    for (int kt = 0; kt < Kc; kt += GBK) {
        const uint4* sa = (const uint4*)(A + (size_t)(m0 + rA) * Kc + kt + hA);
        *(uint4*)&As[rA][hA]     = sa[0];
        *(uint4*)&As[rA][hA + 8] = sa[1];
        *(uint4*)&B1s[rB][qB] = *(const uint4*)(B1T + (size_t)(n0 + rB) * Kc + kt + qB);
        *(uint4*)&B2s[rB][qB] = *(const uint4*)(B2T + (size_t)(n0 + rB) * Kc + kt + qB);
        __syncthreads();

        bf16x16 af0 = load_frag(&As[0][0],  wm * 32,      GBK, 0);
        bf16x16 af1 = load_frag(&As[0][0],  wm * 32 + 16, GBK, 0);
        bf16x16 c10 = load_frag(&B1s[0][0], wn * 32,      GBK, 0);
        bf16x16 c11 = load_frag(&B1s[0][0], wn * 32 + 16, GBK, 0);
        bf16x16 c20 = load_frag(&B2s[0][0], wn * 32,      GBK, 0);
        bf16x16 c21 = load_frag(&B2s[0][0], wn * 32 + 16, GBK, 0);
        acc1[0][0] = wmma_bf16(af0, c10, acc1[0][0]);
        acc1[0][1] = wmma_bf16(af0, c11, acc1[0][1]);
        acc1[1][0] = wmma_bf16(af1, c10, acc1[1][0]);
        acc1[1][1] = wmma_bf16(af1, c11, acc1[1][1]);
        acc2[0][0] = wmma_bf16(af0, c20, acc2[0][0]);
        acc2[0][1] = wmma_bf16(af0, c21, acc2[0][1]);
        acc2[1][0] = wmma_bf16(af1, c20, acc2[1][0]);
        acc2[1][1] = wmma_bf16(af1, c21, acc2[1][1]);
        __syncthreads();
    }

    int hsel = (lane < 16) ? 0 : 8;
    int cl   = lane & 15;
    for (int i = 0; i < 2; i++)
        for (int j = 0; j < 2; j++) {
            int baseRow = m0 + wm * 32 + i * 16 + hsel;
            int baseCol = n0 + wn * 32 + j * 16 + cl;
            float bb1 = b1[baseCol], bb2 = b2[baseCol];
            for (int r = 0; r < 8; r++) {
                size_t off = (size_t)(baseRow + r) * Nc + baseCol;
                float h = acc1[i][j][r] + bb1;
                float g = acc2[i][j][r] + bb2;
                float s = h / (1.f + __expf(-h));     // silu
                outB[off] = f2bf(s * g);
            }
        }
}

// ---------------------------------------------------------------------------
// RoPE in-place on bf16 [N][D] with head-interleaved layout
// ---------------------------------------------------------------------------
__global__ __launch_bounds__(256) void rope_bf16(__bf16* __restrict__ q)
{
    size_t idx = (size_t)blockIdx.x * 256 + threadIdx.x;   // N*NH*(Dh/2)
    int n = (int)(idx / (NHD * (DHD / 2)));
    int rem = (int)(idx % (NHD * (DHD / 2)));
    int h = rem / (DHD / 2);
    int i = rem % (DHD / 2);
    int t = n % TT;
    float inv = __expf(-(2.f * i / (float)DHD) * 9.210340371976184f); // ln 1e4
    float f = (float)t * inv;
    float c = __cosf(f), s = __sinf(f);
    size_t d0 = (size_t)n * DD + h * DHD + 2 * i;
    float x1 = bf2f(q[d0]), x2 = bf2f(q[d0 + 1]);
    q[d0]     = f2bf(x1 * c - x2 * s);
    q[d0 + 1] = f2bf(x1 * s + x2 * c);
}

// ---------------------------------------------------------------------------
// Flash attention: 64 queries / block (4 waves x 16 rows), causal, Dh=64.
// S = Q*K^T and O += P*V both via WMMA; online softmax in fp32.
// ---------------------------------------------------------------------------
#define AQ 64
__global__ __launch_bounds__(128) void flash_attn(
    const __bf16* __restrict__ q, const __bf16* __restrict__ k,
    const __bf16* __restrict__ v, __bf16* __restrict__ ctx)
{
    int bh = blockIdx.y;
    int b = bh >> 4, h = bh & 15;
    int q0 = blockIdx.x * AQ;
    int tid = threadIdx.x, wave = tid >> 5, lane = tid & 31;
    int hsel = (lane < 16) ? 0 : 8;
    int cl   = lane & 15;

    __shared__ __align__(16) __bf16 Ks[32][DHD];
    __shared__ __align__(16) __bf16 Vt[DHD][32];
    __shared__ __align__(16) __bf16 Pw[4][16][32];

    // Q fragments (held in registers for the whole kernel)
    bf16x16 qf0, qf1;
    {
        int qr = q0 + wave * 16 + cl;
        const __bf16* qp = q + (size_t)(b * TT + qr) * DD + h * DHD;
        FragU f0, f1;
        f0.q[0] = *(const uint4*)(qp + hsel);
        f0.q[1] = *(const uint4*)(qp + hsel + 16);
        f1.q[0] = *(const uint4*)(qp + 32 + hsel);
        f1.q[1] = *(const uint4*)(qp + 32 + hsel + 16);
        qf0 = f0.v; qf1 = f1.v;
    }

    float mrow[8], lrow[8];
    for (int r = 0; r < 8; r++) { mrow[r] = -1e30f; lrow[r] = 0.f; }
    floatx8 accO[4] = {};

    int kvEnd = q0 + AQ;                        // causal upper bound
    for (int kv0 = 0; kv0 < kvEnd; kv0 += 32) {
        // Cooperative K / V^T tile loads
        {
            int r = tid >> 2, cb = (tid & 3) * 16;
            const __bf16* kp = k + (size_t)(b * TT + kv0 + r) * DD + h * DHD + cb;
            *(uint4*)&Ks[r][cb]     = *(const uint4*)kp;
            *(uint4*)&Ks[r][cb + 8] = *(const uint4*)(kp + 8);
            const __bf16* vp = v + (size_t)(b * TT + kv0 + r) * DD + h * DHD + cb;
            FragU fv;
            fv.q[0] = *(const uint4*)vp;
            fv.q[1] = *(const uint4*)(vp + 8);
            #pragma unroll
            for (int j = 0; j < 16; j++) Vt[cb + j][r] = fv.v[j];   // transpose
        }
        __syncthreads();

        // S = Q * K^T (two 16x16 tiles covering 32 keys, K-dim = Dh = 2x32)
        floatx8 accS[2];
        #pragma unroll
        for (int j = 0; j < 2; j++) {
            bf16x16 kf0 = load_frag(&Ks[0][0], j * 16, DHD, 0);
            bf16x16 kf1 = load_frag(&Ks[0][0], j * 16, DHD, 32);
            floatx8 z = {};
            z = wmma_bf16(qf0, kf0, z);
            z = wmma_bf16(qf1, kf1, z);
            accS[j] = z;
        }

        // Online softmax (rows live across 16-lane groups)
        float p0v[8], p1v[8];
        #pragma unroll
        for (int r = 0; r < 8; r++) {
            int qrow = q0 + wave * 16 + hsel + r;
            float s0 = accS[0][r] * 0.125f;     // 1/sqrt(64)
            float s1 = accS[1][r] * 0.125f;
            if (kv0 + cl      > qrow) s0 = -1e30f;
            if (kv0 + 16 + cl > qrow) s1 = -1e30f;
            float rm = fmaxf(s0, s1);
            for (int m = 8; m >= 1; m >>= 1) rm = fmaxf(rm, __shfl_xor(rm, m, 32));
            float mn = fmaxf(mrow[r], rm);
            float alpha = __expf(mrow[r] - mn);
            float p0 = __expf(s0 - mn), p1 = __expf(s1 - mn);
            float rs = p0 + p1;
            for (int m = 8; m >= 1; m >>= 1) rs += __shfl_xor(rs, m, 32);
            lrow[r] = lrow[r] * alpha + rs;
            mrow[r] = mn;
            #pragma unroll
            for (int jn = 0; jn < 4; jn++) accO[jn][r] *= alpha;
            p0v[r] = p0; p1v[r] = p1;
        }

        // Re-lay-out P through per-wave LDS so it becomes an A-fragment
        #pragma unroll
        for (int r = 0; r < 8; r++) {
            Pw[wave][hsel + r][cl]      = f2bf(p0v[r]);
            Pw[wave][hsel + r][16 + cl] = f2bf(p1v[r]);
        }
        bf16x16 pf = load_frag(&Pw[wave][0][0], 0, 32, 0);
        #pragma unroll
        for (int jn = 0; jn < 4; jn++) {
            bf16x16 vf = load_frag(&Vt[0][0], jn * 16, 32, 0);
            accO[jn] = wmma_bf16(pf, vf, accO[jn]);
        }
        __syncthreads();
    }

    // Normalize and store ctx (bf16, [b][t][h*64 + dh] layout)
    #pragma unroll
    for (int jn = 0; jn < 4; jn++)
        #pragma unroll
        for (int r = 0; r < 8; r++) {
            int qrow = q0 + wave * 16 + hsel + r;
            float o = accO[jn][r] / lrow[r];
            ctx[(size_t)(b * TT + qrow) * DD + h * DHD + jn * 16 + cl] = f2bf(o);
        }
}

// ---------------------------------------------------------------------------
// Gate: scores = h2 @ gate_w^T (E=8), top-2 softmax -> wts[e][n],
// plus aux-loss accumulators (counts, mean gate probs) via atomics.
// ---------------------------------------------------------------------------
__global__ __launch_bounds__(256) void gate_topk(
    const __bf16* __restrict__ h2, const float* __restrict__ gw,
    float* __restrict__ wts, float* __restrict__ acc)
{
    int n = blockIdx.x;
    const __bf16* xr = h2 + (size_t)n * DD;
    float p[EE] = {};
    for (int d = threadIdx.x; d < DD; d += 256) {
        float vx = bf2f(xr[d]);
        #pragma unroll
        for (int e = 0; e < EE; e++) p[e] += vx * gw[e * DD + d];
    }
    #pragma unroll
    for (int e = 0; e < EE; e++)
        for (int m = 16; m >= 1; m >>= 1) p[e] += __shfl_xor(p[e], m, 32);
    __shared__ float red[8][EE];
    if ((threadIdx.x & 31) == 0)
        for (int e = 0; e < EE; e++) red[threadIdx.x >> 5][e] = p[e];
    __syncthreads();
    if (threadIdx.x == 0) {
        float sc[EE];
        for (int e = 0; e < EE; e++) {
            sc[e] = 0.f;
            for (int w = 0; w < 8; w++) sc[e] += red[w][e];
        }
        int i0 = 0;
        for (int e = 1; e < EE; e++) if (sc[e] > sc[i0]) i0 = e;
        int i1 = -1;
        for (int e = 0; e < EE; e++)
            if (e != i0 && (i1 < 0 || sc[e] > sc[i1])) i1 = e;
        float e0 = 1.f, e1 = __expf(sc[i1] - sc[i0]);
        float inv = 1.f / (e0 + e1);
        for (int e = 0; e < EE; e++)
            wts[(size_t)e * NN + n] = (e == i0) ? e0 * inv
                                    : (e == i1) ? e1 * inv : 0.f;
        atomicAdd(&acc[i0], 1.f);
        atomicAdd(&acc[i1], 1.f);
        float mm = sc[0];
        for (int e = 1; e < EE; e++) mm = fmaxf(mm, sc[e]);
        float pe[EE], ss = 0.f;
        for (int e = 0; e < EE; e++) { pe[e] = __expf(sc[e] - mm); ss += pe[e]; }
        for (int e = 0; e < EE; e++) atomicAdd(&acc[EE + e], pe[e] / ss);
    }
}

__global__ void zero_acc(float* acc) {
    if (threadIdx.x < 2 * EE) acc[threadIdx.x] = 0.f;
}

__global__ void aux_finalize(const float* __restrict__ acc,
                             float* __restrict__ out)
{
    if (threadIdx.x == 0) {
        float s = 0.f;
        for (int e = 0; e < EE; e++)
            s += (acc[e] / (float)NN) * (acc[EE + e] / (float)NN);
        out[0] = (float)EE * s;
    }
}

// ---------------------------------------------------------------------------
// Host-side orchestration
// ---------------------------------------------------------------------------
extern "C" void kernel_launch(void* const* d_in, const int* in_sizes, int n_in,
                              void* d_out, int out_size, void* d_ws, size_t ws_size,
                              hipStream_t stream)
{
    const float* x    = (const float*)d_in[0];
    const float* Wq   = (const float*)d_in[1];
    const float* Wk   = (const float*)d_in[2];
    const float* Wv   = (const float*)d_in[3];
    const float* Wo   = (const float*)d_in[4];
    const float* bo   = (const float*)d_in[5];
    const float* ln1s = (const float*)d_in[6];
    const float* ln1b = (const float*)d_in[7];
    const float* ln2s = (const float*)d_in[8];
    const float* ln2b = (const float*)d_in[9];
    const float* gw   = (const float*)d_in[10];
    const float* w1   = (const float*)d_in[11];
    const float* b1   = (const float*)d_in[12];
    const float* w2   = (const float*)d_in[13];
    const float* b2   = (const float*)d_in[14];
    const float* w3   = (const float*)d_in[15];
    const float* b3   = (const float*)d_in[16];
    float* out = (float*)d_out;

    // Workspace carve-up (~300 MB total)
    char* p = (char*)d_ws;
    auto alloc = [&](size_t bytes) {
        void* r = (void*)p;
        p += (bytes + 255) & ~(size_t)255;
        return r;
    };
    __bf16* h1   = (__bf16*)alloc((size_t)NN * DD * 2);
    __bf16* WqT  = (__bf16*)alloc((size_t)DD * DD * 2);
    __bf16* WkT  = (__bf16*)alloc((size_t)DD * DD * 2);
    __bf16* WvT  = (__bf16*)alloc((size_t)DD * DD * 2);
    __bf16* WoT  = (__bf16*)alloc((size_t)DD * DD * 2);
    __bf16* qb   = (__bf16*)alloc((size_t)NN * DD * 2);
    __bf16* kb   = (__bf16*)alloc((size_t)NN * DD * 2);
    __bf16* vb   = (__bf16*)alloc((size_t)NN * DD * 2);
    __bf16* ctxb = (__bf16*)alloc((size_t)NN * DD * 2);
    float*  x1   = (float*) alloc((size_t)NN * DD * 4);
    __bf16* h2   = (__bf16*)alloc((size_t)NN * DD * 2);
    __bf16* w1T  = (__bf16*)alloc((size_t)EE * DD * HH * 2);
    __bf16* w2T  = (__bf16*)alloc((size_t)EE * DD * HH * 2);
    __bf16* w3T  = (__bf16*)alloc((size_t)EE * HH * DD * 2);
    __bf16* act  = (__bf16*)alloc((size_t)NN * HH * 2);
    float*  wts  = (float*) alloc((size_t)EE * NN * 4);
    float*  accb = (float*) alloc(2 * EE * 4);

    // 1) LN1 -> bf16
    layernorm_bf16<<<NN, 256, 0, stream>>>(x, ln1s, ln1b, h1, DD);

    // 2) Weight convert + transpose (bf16, K-contiguous for WMMA B-frags)
    dim3 cg((DD * DD + 255) / 256, 1, 1);
    cvt_transpose<<<cg, 256, 0, stream>>>(Wq, WqT, DD, DD);
    cvt_transpose<<<cg, 256, 0, stream>>>(Wk, WkT, DD, DD);
    cvt_transpose<<<cg, 256, 0, stream>>>(Wv, WvT, DD, DD);
    cvt_transpose<<<cg, 256, 0, stream>>>(Wo, WoT, DD, DD);
    dim3 cge(((size_t)DD * HH + 255) / 256, 1, EE);
    cvt_transpose<<<cge, 256, 0, stream>>>(w1, w1T, DD, HH);
    cvt_transpose<<<cge, 256, 0, stream>>>(w2, w2T, DD, HH);
    cvt_transpose<<<cge, 256, 0, stream>>>(w3, w3T, HH, DD);

    // 3) Q/K/V projections
    dim3 gg(NN / GBM, DD / GBN);
    gemm_bf16<<<gg, 256, 0, stream>>>(h1, WqT, nullptr, nullptr, nullptr,
                                      nullptr, qb, NN, DD, DD, 0);
    gemm_bf16<<<gg, 256, 0, stream>>>(h1, WkT, nullptr, nullptr, nullptr,
                                      nullptr, kb, NN, DD, DD, 0);
    gemm_bf16<<<gg, 256, 0, stream>>>(h1, WvT, nullptr, nullptr, nullptr,
                                      nullptr, vb, NN, DD, DD, 0);

    // 4) RoPE on q and k
    int ropeBlocks = (NN * NHD * (DHD / 2)) / 256;
    rope_bf16<<<ropeBlocks, 256, 0, stream>>>(qb);
    rope_bf16<<<ropeBlocks, 256, 0, stream>>>(kb);

    // 5) Flash attention -> ctx
    flash_attn<<<dim3(TT / AQ, BB * NHD), 128, 0, stream>>>(qb, kb, vb, ctxb);

    // 6) Output projection + residual: x1 = x + ctx@Wo + bo
    gemm_bf16<<<gg, 256, 0, stream>>>(ctxb, WoT, bo, x, nullptr,
                                      x1, nullptr, NN, DD, DD, 0);
    hipMemcpyAsync(out, x1, (size_t)NN * DD * 4, hipMemcpyDeviceToDevice, stream);

    // 7) LN2 -> bf16, gate + top-2 routing + aux accumulators
    layernorm_bf16<<<NN, 256, 0, stream>>>(x1, ln2s, ln2b, h2, DD);
    zero_acc<<<1, 32, 0, stream>>>(accb);
    gate_topk<<<NN, 256, 0, stream>>>(h2, gw, wts, accb);

    // 8) MoE: per expert, fused up+gate (silu) then scaled down-projection
    //    accumulated into the residual already sitting in d_out.
    for (int e = 0; e < EE; e++) {
        gemm_dual_silu<<<dim3(NN / GBM, HH / GBN), 256, 0, stream>>>(
            h2, w1T + (size_t)e * DD * HH, w2T + (size_t)e * DD * HH,
            b1 + (size_t)e * HH, b2 + (size_t)e * HH,
            wts + (size_t)e * NN, act, NN, HH, DD);
        gemm_bf16<<<dim3(NN / GBM, DD / GBN), 256, 0, stream>>>(
            act, w3T + (size_t)e * HH * DD, b3 + (size_t)e * DD,
            nullptr, wts + (size_t)e * NN,
            out, nullptr, NN, DD, HH, 1);
    }

    // 9) Aux loss scalar
    aux_finalize<<<1, 32, 0, stream>>>(accb, out + (size_t)NN * DD);
}
